// NSA_GPT_1812476199748
// MI455X (gfx1250) — matrix operand order repaired and checked by reading, not compiled, not run on hardware
//
#include <hip/hip_runtime.h>
#include <hip/hip_bf16.h>

// ---------------- model constants ----------------
#define T_    1024
#define D_    768
#define H_    6
#define DH_   128
#define NB_   256
#define CBS_  4
#define NSEL_ 4
#define BLK_  128
#define DFF_  3072
#define VP_   50304
#define NENC_ 6
#define L_    12
#define NBP_  288   // NB+1=257 padded to multiple of 32 (zero-filled)
#define KCROWS_ 384 // kc rows padded to N-tile multiple (zero-filled)

typedef _Float16 v16h __attribute__((ext_vector_type(16)));
typedef _Float16 v8h  __attribute__((ext_vector_type(8)));
typedef float    v8f  __attribute__((ext_vector_type(8)));

static __device__ __forceinline__ v16h cat8(v8h a, v8h b) {
  return __builtin_shufflevector(a, b, 0,1,2,3,4,5,6,7,8,9,10,11,12,13,14,15);
}

// CDNA5 async copy: 16 bytes global -> LDS, tracked by ASYNCcnt (ISA 08 §4)
static __device__ __forceinline__ void async16(unsigned lds_byte, const _Float16* g) {
  asm volatile("global_load_async_to_lds_b128 %0, %1, off"
               :: "v"(lds_byte), "v"(g) : "memory");
}
static __device__ __forceinline__ void wait_async_le4() {
  asm volatile("s_wait_asynccnt 0x4" ::: "memory");
}
static __device__ __forceinline__ void wait_async_0() {
  asm volatile("s_wait_asynccnt 0x0" ::: "memory");
}

// ---------------- batched WMMA GEMM, f16 in / f32 or f16 out ----------------
// C[b] = alpha * (A[b] @ Bt[b]^T)  (+C if accum)  optionally * rowScale[m]
// A: (m,k) at A[m*lda+k].   Bt: (k,n) at Bt[n*ldb+k]  (row n contiguous in k).
// REQUIREMENTS (guaranteed by caller via padding): M,K multiples of tile needs
// covered, Bt has >= ceil(N/128)*128 rows, lda/ldb multiples of 8 halves,
// all pad regions zero-filled. Only the C store is bounds-guarded.
#define TILE_M 128
#define TILE_N 128
#define TILE_K 32
#define ASTR   40   // halves per A-LDS row (80B)
#define BSTR   48   // halves per B-LDS row (96B, 32B-aligned v16h chunks)

__launch_bounds__(256)
__global__ void gemm_wmma(const _Float16* __restrict__ A, int lda, long sAb,
                          const _Float16* __restrict__ Bt, int ldb, long sBb,
                          float* __restrict__ C, _Float16* __restrict__ C16,
                          int ldc, long sCb,
                          int M, int N, int K, float alpha, int accum,
                          const float* __restrict__ rowScale, int rsStride, long rsBatch)
{
  __shared__ __align__(32) _Float16 Alds[2][TILE_M * ASTR];
  __shared__ __align__(32) _Float16 Blds[2][TILE_N * BSTR];

  const int bz = blockIdx.z;
  A  += (long)bz * sAb;
  Bt += (long)bz * sBb;
  C  += (long)bz * sCb;
  if (C16) C16 += (long)bz * sCb;
  const float* rs = rowScale ? rowScale + (long)bz * rsBatch : nullptr;

  const int tid  = threadIdx.x;
  const int m0   = blockIdx.y * TILE_M;
  const int n0   = blockIdx.x * TILE_N;
  const int wave = tid >> 5, lane = tid & 31;
  const int wm   = wave & 1;
  const int wn   = wave >> 1;
  const int c    = lane >> 4;
  const int l16  = lane & 15;

  const _Float16* Ag = A  + (long)m0 * lda;
  const _Float16* Bg = Bt + (long)n0 * ldb;
  const int mmA = tid >> 2, jA = tid & 3;   // 512 16B-chunks per tile, 2/thread

  auto issue = [&](int kt, int bufSel) {
    const int k0 = kt * TILE_K;
    unsigned aB = (unsigned)(unsigned long long)(uintptr_t)&Alds[bufSel][0];
    unsigned bB = (unsigned)(unsigned long long)(uintptr_t)&Blds[bufSel][0];
    async16(aB + mmA * 80 + jA * 16,        Ag + (long)mmA * lda + k0 + jA * 8);
    async16(aB + (mmA + 64) * 80 + jA * 16, Ag + (long)(mmA + 64) * lda + k0 + jA * 8);
    async16(bB + mmA * 96 + jA * 16,        Bg + (long)mmA * ldb + k0 + jA * 8);
    async16(bB + (mmA + 64) * 96 + jA * 16, Bg + (long)(mmA + 64) * ldb + k0 + jA * 8);
  };

  v8f acc[4][2];
  for (int a = 0; a < 4; ++a)
    for (int b = 0; b < 2; ++b)
      acc[a][b] = v8f{0.f,0.f,0.f,0.f,0.f,0.f,0.f,0.f};

  const int nK = (K + TILE_K - 1) / TILE_K;
  issue(0, 0);
  for (int kt = 0; kt < nK; ++kt) {
    const int cur = kt & 1;
    if (kt + 1 < nK) { issue(kt + 1, cur ^ 1); wait_async_le4(); }
    else             { wait_async_0(); }
    __syncthreads();

    v16h bfrag[2];
    for (int b = 0; b < 2; ++b) {
      int n = wn * 32 + b * 16 + l16;
      bfrag[b] = *(const v16h*)&Blds[cur][n * BSTR + c * 16];
    }
    for (int a = 0; a < 4; ++a) {
      int m = wm * 64 + a * 16 + l16;
      const _Float16* ap = &Alds[cur][m * ASTR];
      v8h a0 = *(const v8h*)(ap + c * 16);
      v8h a1 = *(const v8h*)(ap + c * 16 + 8);
      v16h afrag = cat8(a0, a1);
      for (int b = 0; b < 2; ++b)
        acc[a][b] = __builtin_amdgcn_wmma_f32_16x16x32_f16(
            false, afrag, false, bfrag[b], (short)0, acc[a][b], false, false);
    }
    __syncthreads();
  }

  for (int a = 0; a < 4; ++a)
    for (int b = 0; b < 2; ++b) {
      int gn = n0 + wn * 32 + b * 16 + l16;
      if (gn >= N) continue;
      for (int r = 0; r < 8; ++r) {
        int gm = m0 + wm * 64 + a * 16 + r + 8 * c;
        if (gm >= M) continue;
        float val = alpha * acc[a][b][r];
        if (rs) val *= rs[(long)gm * rsStride];
        long o = (long)gm * ldc + gn;
        if (C16) C16[o] = (_Float16)val;
        else if (accum) C[o] += val;
        else C[o] = val;
      }
    }
}

// ---------------- conversion / small kernels ----------------
__global__ void conv16(const float* __restrict__ in, _Float16* __restrict__ out, long n) {
  long i = (long)blockIdx.x * 256 + threadIdx.x;
  if (i < n) out[i] = (_Float16)in[i];
}
// in: batch of [K][N] -> out: batch of [Npad][K], zero-filled for n>=N
__global__ void convT16(const float* __restrict__ in, _Float16* __restrict__ out,
                        int K, int N, int Npad, long total) {
  long i = (long)blockIdx.x * 256 + threadIdx.x;
  if (i >= total) return;
  long per = (long)Npad * K;
  long b = i / per, rem = i % per;
  int n = (int)(rem / K), k = (int)(rem % K);
  out[i] = (n < N) ? (_Float16)in[b * (long)K * N + (long)k * N + n] : (_Float16)0.f;
}
// f16 transpose: in [R][Cc] -> out [Cc][R]
__global__ void transp16(const _Float16* __restrict__ in, _Float16* __restrict__ out,
                         int R, int Cc) {
  long i = (long)blockIdx.x * 256 + threadIdx.x;
  if (i >= (long)R * Cc) return;
  int cc = (int)(i / R), r = (int)(i % R);
  out[i] = in[(long)r * Cc + cc];
}

__global__ void gather_kernel(const int* __restrict__ seq, const float* __restrict__ embed,
                              float* __restrict__ out) {
  int idx = blockIdx.x * 256 + threadIdx.x;
  if (idx >= T_ * D_) return;
  int t = idx / D_, d = idx % D_;
  out[idx] = embed[(long)seq[t] * D_ + d];
}

__global__ void rmsnorm_kernel(const float* __restrict__ x, float* __restrict__ y,
                               _Float16* __restrict__ y16, int cols) {
  __shared__ float red[256];
  int row = blockIdx.x, tid = threadIdx.x;
  const float* xr = x + (long)row * cols;
  float s = 0.f;
  for (int cc = tid; cc < cols; cc += 256) { float v = xr[cc]; s += v * v; }
  red[tid] = s; __syncthreads();
  for (int o = 128; o > 0; o >>= 1) { if (tid < o) red[tid] += red[tid + o]; __syncthreads(); }
  float r = rsqrtf(red[0] / cols + 1e-6f);
  for (int cc = tid; cc < cols; cc += 256) {
    float val = xr[cc] * r;
    if (y)   y[(long)row * cols + cc] = val;
    if (y16) y16[(long)row * cols + cc] = (_Float16)val;
  }
}

__global__ void mix_kernel(float* __restrict__ x, const float* __restrict__ x0,
                           const float* __restrict__ lam, int n) {
  int i = blockIdx.x * 256 + threadIdx.x;
  if (i < n) x[i] = lam[0] * x[i] + lam[1] * x0[i];
}
__global__ void skip_kernel(float* __restrict__ x, const float* __restrict__ s,
                            const float* __restrict__ w, int n) {
  int i = blockIdx.x * 256 + threadIdx.x;
  if (i < n) x[i] += w[0] * s[i];
}
__global__ void sigmoid_kernel(float* __restrict__ g, int n) {
  int i = blockIdx.x * 256 + threadIdx.x;
  if (i < n) g[i] = 1.f / (1.f + __expf(-g[i]));
}
__global__ void relu2_16(const float* __restrict__ h, _Float16* __restrict__ o, int n) {
  int i = blockIdx.x * 256 + threadIdx.x;
  if (i < n) { float v = fmaxf(h[i], 0.f); o[i] = (_Float16)(v * v); }
}
__global__ void zero_kernel(float* p) { p[0] = 0.f; }

// compressed K/V: kc16 [KCROWS_][768] (row n contiguous in d, zero-pad rows),
// vcT16 [768][NBP_] (row d contiguous in block, zero-pad cols)
__global__ void kcvc_kernel(const _Float16* __restrict__ k16, const _Float16* __restrict__ v16,
                            const float* __restrict__ memk, const float* __restrict__ memv,
                            const float* __restrict__ wkp, const float* __restrict__ wvp,
                            _Float16* __restrict__ kc16, _Float16* __restrict__ vcT16) {
  int idx = blockIdx.x * 256 + threadIdx.x;
  if (idx < KCROWS_ * D_) {
    int n = idx / D_, hd = idx % D_;
    float val = 0.f;
    if (n == 0) val = memk[hd];
    else if (n <= NB_) {
      int base = (n - 1) * CBS_;
      for (int cc = 0; cc < CBS_; ++cc) val += wkp[cc] * (float)k16[(long)(base + cc) * D_ + hd];
    }
    kc16[idx] = (_Float16)val;
  }
  if (idx < D_ * NBP_) {
    int d = idx / NBP_, n = idx % NBP_;
    float val = 0.f;
    if (n == 0) val = memv[d];
    else if (n <= NB_) {
      int base = (n - 1) * CBS_;
      for (int cc = 0; cc < CBS_; ++cc) val += wvp[cc] * (float)v16[(long)(base + cc) * D_ + d];
    }
    vcT16[idx] = (_Float16)val;
  }
}

// compressed-branch masked softmax (scf f32 -> pc16 f16) + top-NSEL selection
__global__ void cmp_softmax_topk(const float* __restrict__ scf, _Float16* __restrict__ pc16,
                                 int* __restrict__ sel) {
  int row = blockIdx.x * 256 + threadIdx.x;   // h*T + t
  if (row >= H_ * T_) return;
  int t = row % T_;
  const float* p = scf + (long)row * NBP_;
  _Float16* o = pc16 + (long)row * NBP_;
  int nvalid = t / CBS_;
  float mx = p[0];
  for (int n = 1; n <= nvalid; ++n) mx = fmaxf(mx, p[n]);
  float sum = 0.f;
  for (int n = 0; n <= NB_; ++n)
    sum += (n == 0 || n <= nvalid) ? __expf(p[n] - mx) : 0.f;
  float inv = 1.f / sum;
  float tv[NSEL_]; int ti[NSEL_];
  for (int s2 = 0; s2 < NSEL_; ++s2) { tv[s2] = -1.f; ti[s2] = s2; }
  for (int n = 0; n <= NB_; ++n) {
    float val = (n == 0 || n <= nvalid) ? __expf(p[n] - mx) * inv : 0.f;
    o[n] = (_Float16)val;
    if (n >= 1) {
      float cv = val; int ci = n - 1;
      for (int s2 = 0; s2 < NSEL_; ++s2)
        if (cv > tv[s2]) { float a = tv[s2]; int b = ti[s2]; tv[s2] = cv; ti[s2] = ci; cv = a; ci = b; }
    }
  }
  for (int n = NB_ + 1; n < NBP_; ++n) o[n] = (_Float16)0.f;
  for (int s2 = 0; s2 < NSEL_; ++s2) sel[(long)row * NSEL_ + s2] = ti[s2];
}

// fine (mode 0) / sliding-window (mode 1) masked softmax; wave per row; f16 out
__global__ void attn_softmax(const float* __restrict__ sbuf, _Float16* __restrict__ p16,
                             const int* __restrict__ sel, const int* __restrict__ swnb_p,
                             int use_long, int mode) {
  int gw = (blockIdx.x * blockDim.x + threadIdx.x) >> 5;
  int lane = threadIdx.x & 31;
  if (gw >= H_ * T_) return;
  int t = gw % T_;
  const float* srow = sbuf + (long)gw * T_;
  _Float16* prow = p16 + (long)gw * T_;
  int Wn = 0, s0 = -1, s1 = -1, s2 = -1, s3 = -1;
  if (mode == 1) {
    int nb = swnb_p[0];
    int nb2 = use_long ? nb : ((nb / 2) > 1 ? (nb / 2) : 1);
    Wn = nb2 * BLK_;
  } else {
    const int* sp = sel + (long)gw * NSEL_;
    s0 = sp[0]; s1 = sp[1]; s2 = sp[2]; s3 = sp[3];
  }
  int tb = t >> 2;
  float vals[T_ / 32];
  float mx = -1e30f;
  for (int ii = 0; ii < T_ / 32; ++ii) {
    int ki = lane + ii * 32;
    bool ok;
    if (mode == 1) ok = (ki <= t) && (t - ki < Wn);
    else { int kb = ki >> 2; ok = (ki <= t) && ((kb == tb) || kb == s0 || kb == s1 || kb == s2 || kb == s3); }
    float v = ok ? srow[ki] : -1e30f;
    vals[ii] = v;
    mx = fmaxf(mx, v);
  }
  for (int o = 16; o; o >>= 1) mx = fmaxf(mx, __shfl_xor(mx, o, 32));
  float sum = 0.f;
  for (int ii = 0; ii < T_ / 32; ++ii) {
    float e = (vals[ii] > -1e29f) ? __expf(vals[ii] - mx) : 0.f;
    vals[ii] = e; sum += e;
  }
  for (int o = 16; o; o >>= 1) sum += __shfl_xor(sum, o, 32);
  float inv = 1.f / sum;
  for (int ii = 0; ii < T_ / 32; ++ii)
    prow[lane + ii * 32] = (_Float16)(vals[ii] * inv);
}

// ---------------- fused lm-head + scaled-sigmoid + log-softmax loss ----------------
#define XSTR 784
__launch_bounds__(256)
__global__ void loss_kernel(const _Float16* __restrict__ xn16, const _Float16* __restrict__ lm16,
                            const int* __restrict__ target, float* __restrict__ out) {
  __shared__ __align__(32) _Float16 xl[16 * XSTR];
  __shared__ float ztile[16][128];
  __shared__ float rmax[16], rsum[16], ztgt[16];
  int tid = threadIdx.x;
  int t0 = blockIdx.x * 16;
  for (int idx = tid; idx < 16 * (D_ / 8); idx += 256) {  // 16B chunks
    int r = idx / (D_ / 8), ch = idx % (D_ / 8);
    *(v8h*)&xl[r * XSTR + ch * 8] = *(const v8h*)(xn16 + (long)(t0 + r) * D_ + ch * 8);
  }
  if (tid < 16) { rmax[tid] = -1e30f; rsum[tid] = 0.f; ztgt[tid] = 0.f; }
  __syncthreads();

  int wave = tid >> 5, lane = tid & 31;
  int c = lane >> 4, l16 = lane & 15;
  for (int nb = 0; nb < VP_ / 128; ++nb) {
    int ncol = nb * 128 + wave * 16 + l16;
    v8f acc = v8f{0.f,0.f,0.f,0.f,0.f,0.f,0.f,0.f};
    const _Float16* bp = lm16 + (long)ncol * D_;
    for (int kk = 0; kk < D_ / 32; ++kk) {
      v16h bfrag = *(const v16h*)(bp + kk * 32 + c * 16);
      const _Float16* ap = &xl[l16 * XSTR + kk * 32];
      v8h a0 = *(const v8h*)(ap + c * 16);
      v8h a1 = *(const v8h*)(ap + c * 16 + 8);
      v16h afrag = cat8(a0, a1);
      acc = __builtin_amdgcn_wmma_f32_16x16x32_f16(
          false, afrag, false, bfrag, (short)0, acc, false, false);
    }
    for (int r = 0; r < 8; ++r) {
      float z = 30.f / (1.f + __expf(-acc[r] / 7.5f));
      ztile[r + 8 * c][wave * 16 + l16] = z;
    }
    __syncthreads();
    {
      int row = tid >> 4, cl = tid & 15;
      int tgt = target[t0 + row];
      float lm = -1e30f, tz = 0.f; bool hastz = false;
      float vals[8];
      for (int q2 = 0; q2 < 8; ++q2) {
        int col = cl + q2 * 16;
        float z = ztile[row][col];
        vals[q2] = z;
        lm = fmaxf(lm, z);
        if (nb * 128 + col == tgt) { tz = z; hastz = true; }
      }
      for (int o = 1; o < 16; o <<= 1) lm = fmaxf(lm, __shfl_xor(lm, o, 16));
      float om = rmax[row];
      float nm = fmaxf(om, lm);
      float ls = 0.f;
      for (int q2 = 0; q2 < 8; ++q2) ls += __expf(vals[q2] - nm);
      for (int o = 1; o < 16; o <<= 1) ls += __shfl_xor(ls, o, 16);
      if (cl == 0) { rsum[row] = rsum[row] * __expf(om - nm) + ls; rmax[row] = nm; }
      if (hastz) ztgt[row] = tz;
    }
    __syncthreads();
  }
  if (tid < 16) {
    float loss = (logf(rsum[tid]) + rmax[tid]) - ztgt[tid];
    atomicAdd(out, loss * (1.f / (float)T_));
  }
}

// ---------------- host orchestration ----------------
extern "C" void kernel_launch(void* const* d_in, const int* in_sizes, int n_in,
                              void* d_out, int out_size, void* d_ws, size_t ws_size,
                              hipStream_t stream) {
  const int*   input_seq = (const int*)d_in[0];
  const int*   target    = (const int*)d_in[1];
  const int*   swnb      = (const int*)d_in[2];
  const float* embed     = (const float*)d_in[3];
  const float* wq        = (const float*)d_in[4];
  const float* wk        = (const float*)d_in[5];
  const float* wv        = (const float*)d_in[6];
  const float* wo        = (const float*)d_in[7];
  const float* wg        = (const float*)d_in[8];
  const float* wkp       = (const float*)d_in[9];
  const float* wvp       = (const float*)d_in[10];
  const float* memk      = (const float*)d_in[11];
  const float* memv      = (const float*)d_in[12];
  const float* w1        = (const float*)d_in[13];
  const float* w2        = (const float*)d_in[14];
  const float* lambdas   = (const float*)d_in[15];
  const float* skip_w    = (const float*)d_in[16];
  const float* lm_head   = (const float*)d_in[17];
  float* out = (float*)d_out;
  float* ws  = (float*)d_ws;

  const long XSZ = (long)T_ * D_;             // 786432
  // f32 region
  float* x    = ws;
  float* x0   = x    + XSZ;
  float* attn = x0   + XSZ;
  float* g    = attn + XSZ;                              // T*18
  float* scf  = g    + (long)T_ * H_ * 3;                // H*T*NBP
  float* sbuf = scf  + (long)H_ * T_ * NBP_;             // H*T*T
  float* hbuf = sbuf + (long)H_ * T_ * T_;               // T*DFF (also embed temp)
  float* skips= hbuf + (long)T_ * DFF_;                  // 6*XSZ
  int*   sel  = (int*)(skips + (long)NENC_ * XSZ);       // H*T*NSEL
  // f16 region
  _Float16* f16p = (_Float16*)(sel + (long)H_ * T_ * NSEL_);
  _Float16* xn16  = f16p;            f16p += XSZ;
  _Float16* q16   = f16p;            f16p += XSZ;
  _Float16* k16   = f16p;            f16p += XSZ;
  _Float16* v16   = f16p;            f16p += XSZ;
  _Float16* vT16  = f16p;            f16p += XSZ;
  _Float16* attn16= f16p;            f16p += XSZ;
  _Float16* p16   = f16p;            f16p += (long)H_ * T_ * T_;
  _Float16* pc16  = f16p;            f16p += (long)H_ * T_ * NBP_;
  _Float16* h16   = f16p;            f16p += (long)T_ * DFF_;
  _Float16* kc16  = f16p;            f16p += (long)KCROWS_ * D_;
  _Float16* vcT16 = f16p;            f16p += (long)D_ * NBP_;
  const long DD = (long)D_ * D_;
  const long D1 = (long)D_ * DFF_;
  _Float16* wqT = f16p;              f16p += L_ * DD;
  _Float16* wkT = f16p;              f16p += L_ * DD;
  _Float16* wvT = f16p;              f16p += L_ * DD;
  _Float16* woT = f16p;              f16p += L_ * DD;
  _Float16* wgT = f16p;              f16p += (long)L_ * 128 * D_;
  _Float16* w1T = f16p;              f16p += L_ * D1;
  _Float16* w2T = f16p;              f16p += L_ * D1;
  _Float16* lm16= f16p;              f16p += (long)VP_ * D_;

  const float scale = 0.08838834764831845f;   // 1/sqrt(128)

  auto gemm = [&](const _Float16* A, int lda, long sAb,
                  const _Float16* Bt, int ldb, long sBb,
                  float* C, _Float16* C16, int ldc, long sCb,
                  int M, int N, int K, float alpha, int accum,
                  const float* rs, int rsS, long rsB, int batch) {
    dim3 gr((N + TILE_N - 1) / TILE_N, (M + TILE_M - 1) / TILE_M, batch);
    gemm_wmma<<<gr, 256, 0, stream>>>(A, lda, sAb, Bt, ldb, sBb,
                                      C, C16, ldc, sCb, M, N, K, alpha, accum, rs, rsS, rsB);
  };
  auto blk = [](long n) { return (unsigned)((n + 255) / 256); };
  const int EW = (int)blk(XSZ);
  const int encL[NENC_] = {1, 0, 0, 0, 1, 0};
  const int decL[NENC_] = {0, 1, 0, 0, 0, 1};

  // ---- one-time (per call) weight conversion to f16 (transposed for Bt) ----
  convT16<<<blk(L_ * DD), 256, 0, stream>>>(wq, wqT, D_, D_, D_, L_ * DD);
  convT16<<<blk(L_ * DD), 256, 0, stream>>>(wk, wkT, D_, D_, D_, L_ * DD);
  convT16<<<blk(L_ * DD), 256, 0, stream>>>(wv, wvT, D_, D_, D_, L_ * DD);
  convT16<<<blk(L_ * DD), 256, 0, stream>>>(wo, woT, D_, D_, D_, L_ * DD);
  convT16<<<blk((long)L_ * 128 * D_), 256, 0, stream>>>(wg, wgT, D_, H_ * 3, 128, (long)L_ * 128 * D_);
  convT16<<<blk(L_ * D1), 256, 0, stream>>>(w1, w1T, D_, DFF_, DFF_, L_ * D1);
  convT16<<<blk(L_ * D1), 256, 0, stream>>>(w2, w2T, DFF_, D_, D_, L_ * D1);
  conv16<<<blk((long)VP_ * D_), 256, 0, stream>>>(lm_head, lm16, (long)VP_ * D_);

  // x0 = norm(embed[input_seq]); x = x0
  gather_kernel<<<EW, 256, 0, stream>>>(input_seq, embed, hbuf);
  rmsnorm_kernel<<<T_, 256, 0, stream>>>(hbuf, x0, nullptr, D_);
  hipMemcpyAsync(x, x0, XSZ * sizeof(float), hipMemcpyDeviceToDevice, stream);

  for (int i = 0; i < L_; ++i) {
    if (i >= NENC_) {
      int j = i - NENC_;
      skip_kernel<<<EW, 256, 0, stream>>>(x, skips + (long)(NENC_ - 1 - j) * XSZ,
                                          skip_w + j, (int)XSZ);
    }
    mix_kernel<<<EW, 256, 0, stream>>>(x, x0, lambdas + 2 * i, (int)XSZ);
    rmsnorm_kernel<<<T_, 256, 0, stream>>>(x, nullptr, xn16, D_);

    // projections (f16 outputs for downstream GEMMs)
    gemm(xn16, D_, 0, wqT + i * DD, D_, 0, nullptr, q16, D_, 0, T_, D_, D_, 1.f, 0, nullptr, 0, 0, 1);
    gemm(xn16, D_, 0, wkT + i * DD, D_, 0, nullptr, k16, D_, 0, T_, D_, D_, 1.f, 0, nullptr, 0, 0, 1);
    gemm(xn16, D_, 0, wvT + i * DD, D_, 0, nullptr, v16, D_, 0, T_, D_, D_, 1.f, 0, nullptr, 0, 0, 1);
    gemm(xn16, D_, 0, wgT + (long)i * 128 * D_, D_, 0, g, nullptr, H_ * 3, 0,
         T_, H_ * 3, D_, 1.f, 0, nullptr, 0, 0, 1);
    sigmoid_kernel<<<blk(T_ * H_ * 3), 256, 0, stream>>>(g, T_ * H_ * 3);
    transp16<<<EW, 256, 0, stream>>>(v16, vT16, T_, D_);

    // compressed branch
    kcvc_kernel<<<blk((long)KCROWS_ * D_), 256, 0, stream>>>(
        k16, v16, memk + (long)i * D_, memv + (long)i * D_, wkp + i * CBS_, wvp + i * CBS_,
        kc16, vcT16);
    gemm(q16, D_, DH_, kc16, D_, DH_, scf, nullptr, NBP_, (long)T_ * NBP_,
         T_, NB_ + 1, DH_, scale, 0, nullptr, 0, 0, H_);
    cmp_softmax_topk<<<blk(H_ * T_), 256, 0, stream>>>(scf, pc16, sel);
    gemm(pc16, NBP_, (long)T_ * NBP_, vcT16, NBP_, (long)DH_ * NBP_, attn, nullptr, D_, DH_,
         T_, DH_, NBP_, 1.f, 0, g + 0, H_ * 3, 3, H_);      // attn = g0*out_cmp

    // full scores (shared by fine + sliding window)
    gemm(q16, D_, DH_, k16, D_, DH_, sbuf, nullptr, T_, (long)T_ * T_,
         T_, T_, DH_, scale, 0, nullptr, 0, 0, H_);

    // fine branch
    attn_softmax<<<(H_ * T_) / 8, 256, 0, stream>>>(sbuf, p16, sel, swnb, 0, 0);
    gemm(p16, T_, (long)T_ * T_, vT16, T_, (long)DH_ * T_, attn, nullptr, D_, DH_,
         T_, DH_, T_, 1.f, 1, g + 1, H_ * 3, 3, H_);        // += g1*out_fine

    // sliding-window branch
    int ul = (i < NENC_) ? encL[i] : decL[i - NENC_];
    attn_softmax<<<(H_ * T_) / 8, 256, 0, stream>>>(sbuf, p16, sel, swnb, ul, 1);
    gemm(p16, T_, (long)T_ * T_, vT16, T_, (long)DH_ * T_, attn, nullptr, D_, DH_,
         T_, DH_, T_, 1.f, 1, g + 2, H_ * 3, 3, H_);        // += g2*out_sw

    // output projection: x += attn @ Wo
    conv16<<<EW, 256, 0, stream>>>(attn, attn16, XSZ);
    gemm(attn16, D_, 0, woT + i * DD, D_, 0, x, nullptr, D_, 0, T_, D_, D_, 1.f, 1, nullptr, 0, 0, 1);

    // MLP: x += relu(norm(x)@w1)^2 @ w2
    rmsnorm_kernel<<<T_, 256, 0, stream>>>(x, nullptr, xn16, D_);
    gemm(xn16, D_, 0, w1T + i * D1, D_, 0, hbuf, nullptr, DFF_, 0, T_, DFF_, D_, 1.f, 0, nullptr, 0, 0, 1);
    relu2_16<<<blk((long)T_ * DFF_), 256, 0, stream>>>(hbuf, h16, T_ * DFF_);
    gemm(h16, DFF_, 0, w2T + i * D1, DFF_, 0, x, nullptr, D_, 0, T_, D_, DFF_, 1.f, 1, nullptr, 0, 0, 1);

    if (i < NENC_)
      hipMemcpyAsync(skips + (long)i * XSZ, x, XSZ * sizeof(float),
                     hipMemcpyDeviceToDevice, stream);
  }

  // final norm + fused lm-head loss
  rmsnorm_kernel<<<T_, 256, 0, stream>>>(x, nullptr, xn16, D_);
  zero_kernel<<<1, 1, 0, stream>>>(out);
  loss_kernel<<<T_ / 16, 256, 0, stream>>>(xn16, lm16, target, out);
}
